// InteractionBlock_53833120088087
// MI455X (gfx1250) — matrix operand order repaired and checked by reading, not compiled, not run on hardware
//
#include <hip/hip_runtime.h>

typedef _Float16 F16;
typedef __attribute__((ext_vector_type(16))) _Float16 v16h;
typedef __attribute__((ext_vector_type(8)))  _Float16 v8h;
typedef __attribute__((ext_vector_type(4)))  _Float16 v4h;
typedef __attribute__((ext_vector_type(8)))  float    v8f;

// ---------------------------------------------------------------- utilities

__device__ __forceinline__ float silu_f(float x) {
  // x * sigmoid(x)
  return x * __builtin_amdgcn_rcpf(1.0f + __expf(-x));
}

// copy n halves (n % 8 == 0) from global to LDS, whole block cooperates
__device__ __forceinline__ void cp_lds(F16* dst, const F16* src, int n, int tid, int nthreads) {
  for (int i = tid * 8; i < n; i += nthreads * 8) {
    *(uint4*)(dst + i) = *(const uint4*)(src + i);
  }
}

// stage 16 rows x W f32 columns from global into LDS as f16 (W % 4 == 0)
__device__ __forceinline__ void stage_f32_tile(F16* dst, const float* src, long e0, long E,
                                               int W, int lane) {
  for (int i = lane * 4; i < 16 * W; i += 128) {
    int r = i / W, c = i - r * W;
    long e = e0 + r;
    float4 v = make_float4(0.f, 0.f, 0.f, 0.f);
    if (e < E) v = *(const float4*)(src + e * (long)W + c);
    v4h h; h[0] = (F16)v.x; h[1] = (F16)v.y; h[2] = (F16)v.z; h[3] = (F16)v.w;
    *(v4h*)(dst + r * W + c) = h;
  }
}

// stage 16 rows x W f16 columns from global into LDS (W % 8 == 0)
__device__ __forceinline__ void stage_f16_tile(F16* dst, const F16* src, long e0, long E,
                                               int W, int lane) {
  for (int i = lane * 8; i < 16 * W; i += 256) {
    int r = i / W, c = i - r * W;
    long e = e0 + r;
    uint4 v = make_uint4(0u, 0u, 0u, 0u);
    if (e < E) v = *(const uint4*)(src + e * (long)W + c);
    *(uint4*)(dst + r * W + c) = v;
  }
}

// ------------------------------------------------ wave-level WMMA tile GEMM
// One wave computes C[16 x NT*16] = X[16 x K] * W[K x NT*16].
// X in LDS row-major f16 (row stride = K halves).
// W in LDS as Wt[N][K] (transposed weights, row stride = K halves).
// Accumulators: acc[NT] in f32 via v_wmma_f32_16x16x32_f16.
template <int K, int NT>
__device__ __forceinline__ void wave_gemm(const F16* x, const F16* wt, v8f acc[NT], int lane) {
  constexpr int KC = K / 32;
  const int m = lane & 15, hi = lane >> 4, n = lane & 15;
  v16h a[KC];
#pragma unroll
  for (int kc = 0; kc < KC; ++kc) {
    // 16-bit A 16x32 layout: lanes 0-15 hold K={0..7,16..23}, lanes 16-31 K={8..15,24..31}
    const F16* p = x + m * K + kc * 32 + hi * 8;
    union { v16h v; struct { v8h lo, hi2; } s; } u;
    u.s.lo = *(const v8h*)p;
    u.s.hi2 = *(const v8h*)(p + 16);
    a[kc] = u.v;
  }
#pragma unroll
  for (int nt = 0; nt < NT; ++nt) {
#pragma unroll
    for (int kc = 0; kc < KC; ++kc) {
      // B 32x16: lane (n, hi) holds K = {hi*8..hi*8+7, 16+hi*8..16+hi*8+7} of column n (matches A)
      const F16* q = wt + (nt * 16 + n) * K + kc * 32 + hi * 8;
      union { v16h v; struct { v8h lo, hi2; } s; } u;
      u.s.lo = *(const v8h*)q;
      u.s.hi2 = *(const v8h*)(q + 16);
      acc[nt] = __builtin_amdgcn_wmma_f32_16x16x32_f16(
          false, a[kc], false, u.v, (short)0, acc[nt], false, false);
    }
  }
}

__device__ __forceinline__ void vzero(v8f* a, int n) {
  for (int i = 0; i < n; ++i)
#pragma unroll
    for (int r = 0; r < 8; ++r) a[i][r] = 0.f;
}

// --------------------------------------------- weight convert f32 -> f16^T
__global__ void wconv_k(const float* __restrict__ w, F16* __restrict__ wt, int K, int N) {
  int i = blockIdx.x * blockDim.x + threadIdx.x;
  if (i >= K * N) return;
  int k = i / N, n = i - k * N;
  wt[(size_t)n * K + k] = (F16)w[i];  // Wt[n][k] = W[k][n]
}

// ------------------------------------------------------- kernel 1: edge pre
// m_ang = silu(m_input @ Wkj + b) * ((rbf@rbf1)@rbf2); mad = silu(m_ang @ Wdown)  [E,64] f16
__global__ __launch_bounds__(256) void k1_edge(
    const float* __restrict__ m_input, const float* __restrict__ rbf,
    const float* __restrict__ rbf1, const float* __restrict__ rbf2,
    const float* __restrict__ bkj, const F16* __restrict__ wkj,
    const F16* __restrict__ wdn, F16* __restrict__ mad, int E) {
  __shared__ alignas(16) F16 sWkj[128 * 128];
  __shared__ alignas(16) F16 sWdn[64 * 128];
  __shared__ float sB[128];
  __shared__ float sR1[48];
  __shared__ float sR2[8 * 128];
  __shared__ alignas(16) F16 xb0[8][16 * 128];
  __shared__ alignas(16) F16 xb1[8][16 * 128];
  __shared__ float sT[8][16][8];

  const int tid = threadIdx.x, lane = tid & 31, wv = tid >> 5;
  cp_lds(sWkj, wkj, 128 * 128, tid, 256);
  cp_lds(sWdn, wdn, 64 * 128, tid, 256);
  if (tid < 128) sB[tid] = bkj[tid];
  if (tid < 48) sR1[tid] = rbf1[tid];
  for (int i = tid; i < 1024; i += 256) sR2[i] = rbf2[i];

  const long e0 = (long)blockIdx.x * 128 + wv * 16;
  stage_f32_tile(xb0[wv], m_input, e0, E, 128, lane);
  __syncthreads();

  if (lane < 16) {  // rbf bottleneck: tmp8 = rbf_row @ rbf1
    long e = e0 + lane;
    float rv[6];
#pragma unroll
    for (int i = 0; i < 6; ++i) rv[i] = (e < E) ? rbf[e * 6 + i] : 0.f;
#pragma unroll
    for (int j = 0; j < 8; ++j) {
      float t = 0.f;
#pragma unroll
      for (int i = 0; i < 6; ++i) t += rv[i] * sR1[i * 8 + j];
      sT[wv][lane][j] = t;
    }
  }
  __syncthreads();

  v8f acc[8];
  vzero(acc, 8);
  wave_gemm<128, 8>(xb0[wv], sWkj, acc, lane);

  const int mb = (lane >> 4) * 8, ln = lane & 15;
  float tv[8][8];
#pragma unroll
  for (int r = 0; r < 8; ++r)
#pragma unroll
    for (int j = 0; j < 8; ++j) tv[r][j] = sT[wv][mb + r][j];

#pragma unroll
  for (int nt = 0; nt < 8; ++nt) {
    const int n = nt * 16 + ln;
    const float b = sB[n];
    float rc[8];
#pragma unroll
    for (int j = 0; j < 8; ++j) rc[j] = sR2[j * 128 + n];
#pragma unroll
    for (int r = 0; r < 8; ++r) {
      float rr = 0.f;
#pragma unroll
      for (int j = 0; j < 8; ++j) rr += tv[r][j] * rc[j];
      float v = silu_f(acc[nt][r] + b) * rr;
      xb1[wv][(mb + r) * 128 + n] = (F16)v;
    }
  }
  __syncthreads();

  v8f a2[4];
  vzero(a2, 4);
  wave_gemm<128, 4>(xb1[wv], sWdn, a2, lane);
#pragma unroll
  for (int nt = 0; nt < 4; ++nt) {
    const int n = nt * 16 + ln;
#pragma unroll
    for (int r = 0; r < 8; ++r) {
      long e = e0 + mb + r;
      if (e < E) mad[e * 64 + n] = (F16)silu_f(a2[nt][r]);
    }
  }
}

// ----------------------------------------------------- kernel 2: triplets
// agg[reduce_to_ji] += mad[expand_to_kj] * ((sbf@sbf1)@sbf2)
// reduce_to_ji is sorted -> register accumulation, atomic flush at boundaries.
__global__ __launch_bounds__(256) void k2_triplet(
    const float* __restrict__ sbf, const float* __restrict__ sbf1,
    const float* __restrict__ sbf2, const int* __restrict__ red,
    const int* __restrict__ expd, const F16* __restrict__ mad,
    float* __restrict__ agg, int T) {
  const int lane = threadIdx.x & 31;
  const int wave = (blockIdx.x * blockDim.x + threadIdx.x) >> 5;
  const int nwaves = (gridDim.x * blockDim.x) >> 5;
  const long long chunk = ((long long)T + nwaves - 1) / nwaves;
  long long t0 = (long long)wave * chunk;
  long long t1 = t0 + chunk;
  if (t1 > T) t1 = T;
  if (t0 >= t1) return;

  const int j = lane & 7;
  float s1c[42];
#pragma unroll
  for (int i = 0; i < 42; ++i) s1c[i] = sbf1[i * 8 + j];  // column j of sbf1
  float sbc0[8], sbc1[8];
#pragma unroll
  for (int jj = 0; jj < 8; ++jj) {  // this lane's two sbf2 columns
    sbc0[jj] = sbf2[jj * 64 + 2 * lane];
    sbc1[jj] = sbf2[jj * 64 + 2 * lane + 1];
  }

  float ax = 0.f, ay = 0.f;
  int cur = red[t0];
  for (long long t = t0; t < t1; ++t) {
    const int seg = red[t];  // uniform across wave
    if (seg != cur) {
      unsafeAtomicAdd(agg + (size_t)cur * 64 + 2 * lane, ax);
      unsafeAtomicAdd(agg + (size_t)cur * 64 + 2 * lane + 1, ay);
      ax = 0.f; ay = 0.f; cur = seg;
    }
    const float* srow = sbf + (size_t)t * 42;
    __builtin_prefetch((const void*)(srow + 42 * 8), 0, 0);
    float tj = 0.f;
#pragma unroll
    for (int i = 0; i < 42; ++i) tj += srow[i] * s1c[i];
    float s0 = 0.f, s1v = 0.f;
#pragma unroll
    for (int jj = 0; jj < 8; ++jj) {
      float tb = __shfl(tj, jj, 32);  // broadcast tmp8[jj]
      s0 += tb * sbc0[jj];
      s1v += tb * sbc1[jj];
    }
    const int kj = expd[t];
    const F16* mp = mad + (size_t)kj * 64 + 2 * lane;  // L2-resident gather
    ax += (float)mp[0] * s0;
    ay += (float)mp[1] * s1v;
  }
  unsafeAtomicAdd(agg + (size_t)cur * 64 + 2 * lane, ax);
  unsafeAtomicAdd(agg + (size_t)cur * 64 + 2 * lane + 1, ay);
}

// --------------------- kernel 3A: up-proj + dense_ji + res_before (fused)
// m  = silu(agg @ Wup) + silu(m_input @ Wji + bji)
// m  = m + silu(silu(m@w1+b1)@w2+b2)        -> mmid [E,128] f16
__global__ __launch_bounds__(128) void k3A(
    const float* __restrict__ m_input, const float* __restrict__ agg,
    const F16* __restrict__ wup, const F16* __restrict__ wji,
    const float* __restrict__ bji,
    const F16* __restrict__ w1, const float* __restrict__ b1,
    const F16* __restrict__ w2, const float* __restrict__ b2,
    F16* __restrict__ mmid, int E) {
  __shared__ alignas(16) F16 sWup[128 * 64];
  __shared__ alignas(16) F16 sWji[128 * 128];
  __shared__ alignas(16) F16 sW1[128 * 128];
  __shared__ alignas(16) F16 sW2[128 * 128];
  __shared__ float sBji[128], sB1[128], sB2[128];
  __shared__ alignas(16) F16 xb0[4][16 * 128];
  __shared__ alignas(16) F16 xb1[4][16 * 128];
  __shared__ alignas(16) F16 xbA[4][16 * 64];

  const int tid = threadIdx.x, lane = tid & 31, wv = tid >> 5;
  cp_lds(sWup, wup, 128 * 64, tid, 128);
  cp_lds(sWji, wji, 128 * 128, tid, 128);
  cp_lds(sW1, w1, 128 * 128, tid, 128);
  cp_lds(sW2, w2, 128 * 128, tid, 128);
  if (tid < 128) { sBji[tid] = bji[tid]; sB1[tid] = b1[tid]; sB2[tid] = b2[tid]; }

  const long e0 = (long)blockIdx.x * 64 + wv * 16;
  stage_f32_tile(xbA[wv], agg, e0, E, 64, lane);
  stage_f32_tile(xb0[wv], m_input, e0, E, 128, lane);
  __syncthreads();

  const int mb = (lane >> 4) * 8, ln = lane & 15;
  v8f acc[8];
  float mf[8][8];

  // propagated = silu(agg @ Wup)
  vzero(acc, 8);
  wave_gemm<64, 8>(xbA[wv], sWup, acc, lane);
#pragma unroll
  for (int nt = 0; nt < 8; ++nt)
#pragma unroll
    for (int r = 0; r < 8; ++r) mf[nt][r] = silu_f(acc[nt][r]);

  // m = silu(m_input @ Wji + bji) + propagated
  vzero(acc, 8);
  wave_gemm<128, 8>(xb0[wv], sWji, acc, lane);
#pragma unroll
  for (int nt = 0; nt < 8; ++nt) {
    const int n = nt * 16 + ln;
    const float b = sBji[n];
#pragma unroll
    for (int r = 0; r < 8; ++r) {
      mf[nt][r] = silu_f(acc[nt][r] + b) + mf[nt][r];
      xb0[wv][(mb + r) * 128 + n] = (F16)mf[nt][r];
    }
  }
  __syncthreads();

  // res_before
  vzero(acc, 8);
  wave_gemm<128, 8>(xb0[wv], sW1, acc, lane);
#pragma unroll
  for (int nt = 0; nt < 8; ++nt) {
    const int n = nt * 16 + ln;
    const float b = sB1[n];
#pragma unroll
    for (int r = 0; r < 8; ++r) xb1[wv][(mb + r) * 128 + n] = (F16)silu_f(acc[nt][r] + b);
  }
  __syncthreads();

  vzero(acc, 8);
  wave_gemm<128, 8>(xb1[wv], sW2, acc, lane);
#pragma unroll
  for (int nt = 0; nt < 8; ++nt) {
    const int n = nt * 16 + ln;
    const float b = sB2[n];
#pragma unroll
    for (int r = 0; r < 8; ++r) {
      float v = mf[nt][r] + silu_f(acc[nt][r] + b);
      long e = e0 + mb + r;
      if (e < E) mmid[e * 128 + n] = (F16)v;
    }
  }
}

// ---------------------- kernel 3B: final dense + m_input skip + res_after[0]
// m = silu(mmid @ Wf + bf) + m_input; m = m + silu(silu(m@w1+b1)@w2+b2) -> mmid
__global__ __launch_bounds__(128) void k3B(
    const float* __restrict__ m_input, const F16* __restrict__ wf,
    const float* __restrict__ bf,
    const F16* __restrict__ w1, const float* __restrict__ b1,
    const F16* __restrict__ w2, const float* __restrict__ b2,
    F16* __restrict__ mio, int E) {
  __shared__ alignas(16) F16 sWf[128 * 128];
  __shared__ alignas(16) F16 sW1[128 * 128];
  __shared__ alignas(16) F16 sW2[128 * 128];
  __shared__ float sBf[128], sB1[128], sB2[128];
  __shared__ alignas(16) F16 xb0[4][16 * 128];
  __shared__ alignas(16) F16 xb1[4][16 * 128];

  const int tid = threadIdx.x, lane = tid & 31, wv = tid >> 5;
  cp_lds(sWf, wf, 128 * 128, tid, 128);
  cp_lds(sW1, w1, 128 * 128, tid, 128);
  cp_lds(sW2, w2, 128 * 128, tid, 128);
  if (tid < 128) { sBf[tid] = bf[tid]; sB1[tid] = b1[tid]; sB2[tid] = b2[tid]; }

  const long e0 = (long)blockIdx.x * 64 + wv * 16;
  stage_f16_tile(xb0[wv], mio, e0, E, 128, lane);
  __syncthreads();

  const int mb = (lane >> 4) * 8, ln = lane & 15;
  v8f acc[8];
  float mf[8][8];

  // m = silu(x @ Wf + bf) + m_input
  vzero(acc, 8);
  wave_gemm<128, 8>(xb0[wv], sWf, acc, lane);
#pragma unroll
  for (int nt = 0; nt < 8; ++nt) {
    const int n = nt * 16 + ln;
    const float b = sBf[n];
#pragma unroll
    for (int r = 0; r < 8; ++r) {
      long e = e0 + mb + r;
      float skip = (e < E) ? m_input[e * 128 + n] : 0.f;
      mf[nt][r] = silu_f(acc[nt][r] + b) + skip;
    }
  }
  __syncthreads();  // xb0 reads done; safe to overwrite
#pragma unroll
  for (int nt = 0; nt < 8; ++nt) {
    const int n = nt * 16 + ln;
#pragma unroll
    for (int r = 0; r < 8; ++r) xb0[wv][(mb + r) * 128 + n] = (F16)mf[nt][r];
  }
  __syncthreads();

  // res_after[0]
  vzero(acc, 8);
  wave_gemm<128, 8>(xb0[wv], sW1, acc, lane);
#pragma unroll
  for (int nt = 0; nt < 8; ++nt) {
    const int n = nt * 16 + ln;
    const float b = sB1[n];
#pragma unroll
    for (int r = 0; r < 8; ++r) xb1[wv][(mb + r) * 128 + n] = (F16)silu_f(acc[nt][r] + b);
  }
  __syncthreads();

  vzero(acc, 8);
  wave_gemm<128, 8>(xb1[wv], sW2, acc, lane);
#pragma unroll
  for (int nt = 0; nt < 8; ++nt) {
    const int n = nt * 16 + ln;
    const float b = sB2[n];
#pragma unroll
    for (int r = 0; r < 8; ++r) {
      float v = mf[nt][r] + silu_f(acc[nt][r] + b);
      long e = e0 + mb + r;
      if (e < E) mio[e * 128 + n] = (F16)v;
    }
  }
}

// ------------------------------------ residual block, writes f32 output
// out = x + silu(silu(x@w1+b1)@w2+b2)
__global__ __launch_bounds__(128) void kres(
    const F16* __restrict__ w1, const float* __restrict__ b1,
    const F16* __restrict__ w2, const float* __restrict__ b2,
    const F16* __restrict__ mio, float* __restrict__ out_f32, int E) {
  __shared__ alignas(16) F16 sW1[128 * 128];
  __shared__ alignas(16) F16 sW2[128 * 128];
  __shared__ float sB1[128], sB2[128];
  __shared__ alignas(16) F16 xb0[4][16 * 128];
  __shared__ alignas(16) F16 xb1[4][16 * 128];

  const int tid = threadIdx.x, lane = tid & 31, wv = tid >> 5;
  cp_lds(sW1, w1, 128 * 128, tid, 128);
  cp_lds(sW2, w2, 128 * 128, tid, 128);
  if (tid < 128) { sB1[tid] = b1[tid]; sB2[tid] = b2[tid]; }

  const long e0 = (long)blockIdx.x * 64 + wv * 16;
  stage_f16_tile(xb0[wv], mio, e0, E, 128, lane);
  __syncthreads();

  const int mb = (lane >> 4) * 8, ln = lane & 15;
  float mf[8][8];  // this lane's skip values
#pragma unroll
  for (int nt = 0; nt < 8; ++nt)
#pragma unroll
    for (int r = 0; r < 8; ++r) mf[nt][r] = (float)xb0[wv][(mb + r) * 128 + nt * 16 + ln];

  v8f acc[8];
  vzero(acc, 8);
  wave_gemm<128, 8>(xb0[wv], sW1, acc, lane);
#pragma unroll
  for (int nt = 0; nt < 8; ++nt) {
    const int n = nt * 16 + ln;
    const float b = sB1[n];
#pragma unroll
    for (int r = 0; r < 8; ++r) xb1[wv][(mb + r) * 128 + n] = (F16)silu_f(acc[nt][r] + b);
  }
  __syncthreads();

  vzero(acc, 8);
  wave_gemm<128, 8>(xb1[wv], sW2, acc, lane);
#pragma unroll
  for (int nt = 0; nt < 8; ++nt) {
    const int n = nt * 16 + ln;
    const float b = sB2[n];
#pragma unroll
    for (int r = 0; r < 8; ++r) {
      float v = mf[nt][r] + silu_f(acc[nt][r] + b);
      long e = e0 + mb + r;
      if (e < E) out_f32[e * 128 + n] = v;
    }
  }
}

// ------------------------------------------------------------------- host
extern "C" void kernel_launch(void* const* d_in, const int* in_sizes, int n_in,
                              void* d_out, int out_size, void* d_ws, size_t ws_size,
                              hipStream_t stream) {
  // setup_inputs() insertion-order flattening
  const float* m_input = (const float*)d_in[0];
  const float* rbf     = (const float*)d_in[1];
  const float* sbf     = (const float*)d_in[2];
  const float* rbf1    = (const float*)d_in[3];
  const float* rbf2    = (const float*)d_in[4];
  const float* sbf1    = (const float*)d_in[5];
  const float* sbf2    = (const float*)d_in[6];
  const float* w_kj    = (const float*)d_in[7];
  const float* b_kj    = (const float*)d_in[8];
  const float* w_dn    = (const float*)d_in[9];
  const float* w_up    = (const float*)d_in[10];
  const float* w_ji    = (const float*)d_in[11];
  const float* b_ji    = (const float*)d_in[12];
  const float* rb_w1   = (const float*)d_in[13];
  const float* rb_b1   = (const float*)d_in[14];
  const float* rb_w2   = (const float*)d_in[15];
  const float* rb_b2   = (const float*)d_in[16];
  const float* fin_w   = (const float*)d_in[17];
  const float* fin_b   = (const float*)d_in[18];
  const float* a0_w1   = (const float*)d_in[19];
  const float* a0_b1   = (const float*)d_in[20];
  const float* a0_w2   = (const float*)d_in[21];
  const float* a0_b2   = (const float*)d_in[22];
  const float* a1_w1   = (const float*)d_in[23];
  const float* a1_b1   = (const float*)d_in[24];
  const float* a1_w2   = (const float*)d_in[25];
  const float* a1_b2   = (const float*)d_in[26];
  const int* red       = (const int*)d_in[27];
  const int* expd      = (const int*)d_in[28];

  const int E = in_sizes[0] / 128;
  const int T = in_sizes[27];

  char* ws = (char*)d_ws;
  size_t off = 0;
  auto alloc = [&](size_t bytes) -> char* {
    char* p = ws + off;
    off = (off + bytes + 255) & ~(size_t)255;
    return p;
  };
  F16* h_kj   = (F16*)alloc(128 * 128 * 2);
  F16* h_dn   = (F16*)alloc(64 * 128 * 2);
  F16* h_up   = (F16*)alloc(128 * 64 * 2);
  F16* h_ji   = (F16*)alloc(128 * 128 * 2);
  F16* h_r1   = (F16*)alloc(128 * 128 * 2);
  F16* h_r2   = (F16*)alloc(128 * 128 * 2);
  F16* h_fw   = (F16*)alloc(128 * 128 * 2);
  F16* h_a0w1 = (F16*)alloc(128 * 128 * 2);
  F16* h_a0w2 = (F16*)alloc(128 * 128 * 2);
  F16* h_a1w1 = (F16*)alloc(128 * 128 * 2);
  F16* h_a1w2 = (F16*)alloc(128 * 128 * 2);
  F16*   mad  = (F16*)alloc((size_t)E * 64 * sizeof(F16));
  float* agg  = (float*)alloc((size_t)E * 64 * sizeof(float));
  F16*   mmid = (F16*)alloc((size_t)E * 128 * sizeof(F16));
  (void)ws_size; (void)n_in; (void)out_size;

  auto wc = [&](const float* s, F16* d, int K, int N) {
    int n = K * N;
    wconv_k<<<dim3((n + 255) / 256), dim3(256), 0, stream>>>(s, d, K, N);
  };
  wc(w_kj, h_kj, 128, 128);
  wc(w_dn, h_dn, 128, 64);
  wc(w_up, h_up, 64, 128);
  wc(w_ji, h_ji, 128, 128);
  wc(rb_w1, h_r1, 128, 128);
  wc(rb_w2, h_r2, 128, 128);
  wc(fin_w, h_fw, 128, 128);
  wc(a0_w1, h_a0w1, 128, 128);
  wc(a0_w2, h_a0w2, 128, 128);
  wc(a1_w1, h_a1w1, 128, 128);
  wc(a1_w2, h_a1w2, 128, 128);

  hipMemsetAsync(agg, 0, (size_t)E * 64 * sizeof(float), stream);

  k1_edge<<<dim3((E + 127) / 128), dim3(256), 0, stream>>>(
      m_input, rbf, rbf1, rbf2, b_kj, h_kj, h_dn, mad, E);

  k2_triplet<<<dim3(1024), dim3(256), 0, stream>>>(
      sbf, sbf1, sbf2, red, expd, mad, agg, T);

  const dim3 g64((E + 63) / 64), b128(128);
  k3A<<<g64, b128, 0, stream>>>(m_input, agg, h_up, h_ji, b_ji,
                                h_r1, rb_b1, h_r2, rb_b2, mmid, E);
  k3B<<<g64, b128, 0, stream>>>(m_input, h_fw, fin_b,
                                h_a0w1, a0_b1, h_a0w2, a0_b2, mmid, E);
  kres<<<g64, b128, 0, stream>>>(h_a1w1, a1_b1, h_a1w2, a1_b2, mmid,
                                 (float*)d_out, E);
}
